// BinaryAssociativeMemory_51470888075839
// MI455X (gfx1250) — matrix operand order repaired and checked by reading, not compiled, not run on hardware
//
#include <hip/hip_runtime.h>

typedef __bf16 bf16;
typedef __attribute__((ext_vector_type(8)))  __bf16 v8bf;
typedef __attribute__((ext_vector_type(16))) __bf16 v16bf;
typedef __attribute__((ext_vector_type(8)))  float  v8f;
typedef __attribute__((ext_vector_type(4)))  float  v4f;
typedef __attribute__((ext_vector_type(4)))  int    v4i;

#define WMMA_BF16(a, b, c) \
  __builtin_amdgcn_wmma_f32_16x16x32_bf16(false, (a), false, (b), (short)0, (c), false, false)

static __device__ __forceinline__ v16bf cat8(v8bf lo, v8bf hi) {
  return __builtin_shufflevector(lo, hi, 0,1,2,3,4,5,6,7,8,9,10,11,12,13,14,15);
}

// ---- CDNA5 async global->LDS copy (ASYNCcnt path), with safe fallback ----
// Signature (from hipcc diagnostic): (int4 AS1*, int4 AS3*, imm offset, imm cpol)
#if __has_builtin(__builtin_amdgcn_global_load_async_to_lds_b128)
#define ASYNC_OK 1
#define ASYNC_CP16(gp, lp)                                                  \
  __builtin_amdgcn_global_load_async_to_lds_b128(                           \
      (__attribute__((address_space(1))) v4i*)(gp),                         \
      (__attribute__((address_space(3))) v4i*)(lp), 0, 0)
#else
#define ASYNC_OK 0
#endif

static __device__ __forceinline__ void wait_async0() {
#if __has_builtin(__builtin_amdgcn_s_wait_asynccnt)
  __builtin_amdgcn_s_wait_asynccnt(0);
#else
  asm volatile("s_wait_asynccnt 0x0" ::: "memory");
#endif
}

// Problem constants
constexpr int Bb = 2, Tt = 4096, Dd = 1024, Hh = 16, DH = 64, Cc = 64, NC = 64;
constexpr int MROWS = Bb * Tt;               // 8192
constexpr float QSCALE = 0.125f;             // 1/sqrt(64)

// ---------------------------------------------------------------------------
// Kernel 1: QKV GEMM  (M=8192, N=3072, K=1024), fp32 in, bf16 WMMA core.
// Epilogue: scatter into q/k/v buffers laid out (B,H,T,d) bf16, with
// q *= 1/8, k = sign(k), v = sign(v).
// ---------------------------------------------------------------------------
__global__ __launch_bounds__(256)
void qkv_gemm_kernel(const float* __restrict__ X, const float* __restrict__ W,
                     bf16* __restrict__ q_ws, bf16* __restrict__ k_ws,
                     bf16* __restrict__ v_ws) {
  __shared__ bf16 As[128][40];
  __shared__ bf16 Bs[128][40];
  const int n0 = blockIdx.x * 128;
  const int m0 = blockIdx.y * 128;
  const int tid  = threadIdx.x;
  const int lane = tid & 31, wave = tid >> 5;
  const int wm = (wave & 3) * 32;        // 4 waves along M
  const int wn = (wave >> 2) * 64;       // 2 waves along N
  const int lr = lane & 15, lh = lane >> 4;

  v8f acc[2][4] = {};

  for (int kt = 0; kt < 1024; kt += 32) {
    const int ktn = (kt + 32 < 1024) ? (kt + 32) : kt;   // clamped, branch-free
    // ---- global fp32 -> bf16 LDS (A tile 128x32, B^T tile 128x32) ----
#pragma unroll
    for (int c = 0; c < 4; ++c) {
      int q   = tid + 256 * c;                  // 0..1023 float4 chunks
      int row = q >> 3, col = (q & 7) * 4;
      const float* ap = X + (size_t)(m0 + row) * 1024 + kt + col;
      const float* bp = W + (size_t)(n0 + row) * 1024 + kt + col;
      __builtin_prefetch(X + (size_t)(m0 + row) * 1024 + ktn + col, 0, 3);
      __builtin_prefetch(W + (size_t)(n0 + row) * 1024 + ktn + col, 0, 3);
      v4f a4 = *(const v4f*)ap;
      v4f b4 = *(const v4f*)bp;
#pragma unroll
      for (int j = 0; j < 4; ++j) {
        As[row][col + j] = (bf16)a4[j];
        Bs[row][col + j] = (bf16)b4[j];
      }
    }
    __syncthreads();

    // ---- fragments + WMMA ----
    v16bf af[2], bfr[4];
#pragma unroll
    for (int mt = 0; mt < 2; ++mt) {
      const bf16* ap = &As[wm + mt * 16 + lr][0];
      af[mt] = cat8(*(const v8bf*)(ap + 8 * lh), *(const v8bf*)(ap + 16 + 8 * lh));
    }
#pragma unroll
    for (int nt = 0; nt < 4; ++nt) {
      const bf16* bp = &Bs[wn + nt * 16 + lr][0];
      bfr[nt] = cat8(*(const v8bf*)(bp + 8 * lh), *(const v8bf*)(bp + 16 + 8 * lh));
    }
#pragma unroll
    for (int mt = 0; mt < 2; ++mt)
#pragma unroll
      for (int nt = 0; nt < 4; ++nt)
        acc[mt][nt] = WMMA_BF16(af[mt], bfr[nt], acc[mt][nt]);
    __syncthreads();
  }

  // ---- epilogue: scatter to (B,H,T,d) with scale / binarize ----
#pragma unroll
  for (int nt = 0; nt < 4; ++nt) {
    int gn  = n0 + wn + nt * 16 + lr;          // 0..3071
    int s   = gn >> 10;                        // 0=q 1=k 2=v
    int rem = gn & 1023;
    int hh  = rem >> 6, dd = rem & 63;
    bf16* dst = (s == 0) ? q_ws : (s == 1) ? k_ws : v_ws;
#pragma unroll
    for (int mt = 0; mt < 2; ++mt) {
#pragma unroll
      for (int r = 0; r < 8; ++r) {
        int gm = m0 + wm + mt * 16 + 8 * lh + r;
        int b  = gm >> 12, t = gm & 4095;
        size_t idx = (((size_t)(b * Hh + hh)) * Tt + t) * DH + dd;
        float val = acc[mt][nt][r];
        if (s == 0) dst[idx] = (bf16)(val * QSCALE);
        else        dst[idx] = (bf16)((val >= 0.f) ? 1.f : -1.f);
      }
    }
  }
}

// ---------------------------------------------------------------------------
// Kernel 2: chunk_kv[b,h,n] = Kc^T @ Vc   (64x64, K over chunk positions)
// ---------------------------------------------------------------------------
__global__ __launch_bounds__(128)
void chunkkv_kernel(const bf16* __restrict__ k_ws, const bf16* __restrict__ v_ws,
                    float* __restrict__ ckv) {
  __shared__ bf16 Kt[64][72];
  __shared__ bf16 Vt[64][72];
  const int n = blockIdx.x, bh = blockIdx.y;
  const size_t base = ((size_t)bh * Tt + n * Cc) * DH;
  const int tid = threadIdx.x;

#pragma unroll
  for (int c = 0; c < 4; ++c) {
    int q   = tid + 128 * c;                 // 0..511 v8 chunks
    int row = q >> 3, col = (q & 7) * 8;
    v8bf kv = *(const v8bf*)(k_ws + base + row * 64 + col);
    v8bf vv = *(const v8bf*)(v_ws + base + row * 64 + col);
#pragma unroll
    for (int j = 0; j < 8; ++j) { Kt[col + j][row] = kv[j]; Vt[col + j][row] = vv[j]; }
  }
  __syncthreads();

  const int lane = tid & 31, wave = tid >> 5;
  const int lr = lane & 15, lh = lane >> 4;
  v8f acc[4] = {};
#pragma unroll
  for (int ks = 0; ks < 64; ks += 32) {
    const bf16* ap = &Kt[wave * 16 + lr][ks];
    v16bf a = cat8(*(const v8bf*)(ap + 8 * lh), *(const v8bf*)(ap + 16 + 8 * lh));
#pragma unroll
    for (int nt = 0; nt < 4; ++nt) {
      const bf16* bp = &Vt[nt * 16 + lr][ks];
      v16bf b = cat8(*(const v8bf*)(bp + 8 * lh), *(const v8bf*)(bp + 16 + 8 * lh));
      acc[nt] = WMMA_BF16(a, b, acc[nt]);
    }
  }
  float* out = ckv + ((size_t)bh * NC + n) * 4096;
#pragma unroll
  for (int nt = 0; nt < 4; ++nt)
#pragma unroll
    for (int r = 0; r < 8; ++r)
      out[(wave * 16 + 8 * lh + r) * 64 + nt * 16 + lr] = acc[nt][r];
}

// ---------------------------------------------------------------------------
// Kernel 3: exclusive prefix over chunks + final_state
// ---------------------------------------------------------------------------
__global__ __launch_bounds__(256)
void prefix_kernel(const float* __restrict__ ckv, float* __restrict__ pre,
                   float* __restrict__ fstate) {
  const int bh = blockIdx.x;
  const int e0 = threadIdx.x * 16;
  const float* src = ckv + (size_t)bh * NC * 4096;
  float*       dst = pre + (size_t)bh * NC * 4096;
  v4f run[4] = {};
  for (int n = 0; n < NC; ++n) {
    const float* s = src + n * 4096 + e0;
    float*       d = dst + n * 4096 + e0;
#pragma unroll
    for (int j = 0; j < 4; ++j) {
      *(v4f*)(d + j * 4) = run[j];
      run[j] += *(const v4f*)(s + j * 4);
    }
  }
  float* f = fstate + (size_t)bh * 4096 + e0;
#pragma unroll
  for (int j = 0; j < 4; ++j) *(v4f*)(f + j * 4) = run[j];
}

// ---------------------------------------------------------------------------
// Kernel 4: per (b,h,chunk):  S = Q K^T (masked)  ;  O = S V + Q prefix
// stored to attn buffer in (B, T, H*d) layout (bf16)
// ---------------------------------------------------------------------------
__global__ __launch_bounds__(128)
void attn_out_kernel(const bf16* __restrict__ q_ws, const bf16* __restrict__ k_ws,
                     const bf16* __restrict__ v_ws, const float* __restrict__ pre,
                     bf16* __restrict__ attn) {
  __shared__ bf16 Qs[64][72];
  __shared__ bf16 Ks[64][72];
  __shared__ bf16 Vt[64][72];
  __shared__ bf16 Pt[64][72];
  __shared__ bf16 Sb[64][72];
  const int n = blockIdx.x, bh = blockIdx.y;
  const int b = bh >> 4, h = bh & 15;
  const size_t base = ((size_t)bh * Tt + n * Cc) * DH;
  const int tid = threadIdx.x;

  // Q and K are verbatim copies: use async global->LDS (ASYNCcnt) when available
#if ASYNC_OK
#pragma unroll
  for (int c = 0; c < 4; ++c) {
    int q   = tid + 128 * c;                 // 0..511 v8 chunks
    int row = q >> 3, col = (q & 7) * 8;
    ASYNC_CP16(q_ws + base + row * 64 + col, &Qs[row][col]);
    ASYNC_CP16(k_ws + base + row * 64 + col, &Ks[row][col]);
  }
#endif
#pragma unroll
  for (int c = 0; c < 4; ++c) {
    int q   = tid + 128 * c;
    int row = q >> 3, col = (q & 7) * 8;
#if !ASYNC_OK
    v8bf qv = *(const v8bf*)(q_ws + base + row * 64 + col);
    v8bf kv = *(const v8bf*)(k_ws + base + row * 64 + col);
    *(v8bf*)&Qs[row][col] = qv;
    *(v8bf*)&Ks[row][col] = kv;
#endif
    v8bf vv = *(const v8bf*)(v_ws + base + row * 64 + col);
#pragma unroll
    for (int j = 0; j < 8; ++j) Vt[col + j][row] = vv[j];
  }
  const float* P = pre + ((size_t)bh * NC + n) * 4096;
#pragma unroll
  for (int c = 0; c < 8; ++c) {
    int q   = tid + 128 * c;                 // 0..1023 float4 chunks
    int row = q >> 4, col = (q & 15) * 4;
    v4f pv = *(const v4f*)(P + row * 64 + col);
#pragma unroll
    for (int j = 0; j < 4; ++j) Pt[col + j][row] = (bf16)pv[j];
  }
#if ASYNC_OK
  wait_async0();
#endif
  __syncthreads();

  const int lane = tid & 31, wave = tid >> 5;
  const int lr = lane & 15, lh = lane >> 4;

  // Phase 1: masked scores -> bf16
  {
    v8f sacc[4] = {};
#pragma unroll
    for (int ks = 0; ks < 64; ks += 32) {
      const bf16* ap = &Qs[wave * 16 + lr][ks];
      v16bf a = cat8(*(const v8bf*)(ap + 8 * lh), *(const v8bf*)(ap + 16 + 8 * lh));
#pragma unroll
      for (int nt = 0; nt < 4; ++nt) {
        const bf16* bp = &Ks[nt * 16 + lr][ks];
        v16bf bb = cat8(*(const v8bf*)(bp + 8 * lh), *(const v8bf*)(bp + 16 + 8 * lh));
        sacc[nt] = WMMA_BF16(a, bb, sacc[nt]);
      }
    }
#pragma unroll
    for (int nt = 0; nt < 4; ++nt) {
      int j = nt * 16 + lr;
#pragma unroll
      for (int r = 0; r < 8; ++r) {
        int i = wave * 16 + 8 * lh + r;
        float v = (j <= i) ? sacc[nt][r] : 0.f;   // causal mask
        Sb[i][j] = (bf16)v;
      }
    }
  }
  __syncthreads();

  // Phase 2: O = S V + Q P (same accumulators)
  v8f oacc[4] = {};
#pragma unroll
  for (int ks = 0; ks < 64; ks += 32) {
    const bf16* ap = &Sb[wave * 16 + lr][ks];
    v16bf a = cat8(*(const v8bf*)(ap + 8 * lh), *(const v8bf*)(ap + 16 + 8 * lh));
#pragma unroll
    for (int nt = 0; nt < 4; ++nt) {
      const bf16* bp = &Vt[nt * 16 + lr][ks];
      v16bf bb = cat8(*(const v8bf*)(bp + 8 * lh), *(const v8bf*)(bp + 16 + 8 * lh));
      oacc[nt] = WMMA_BF16(a, bb, oacc[nt]);
    }
  }
#pragma unroll
  for (int ks = 0; ks < 64; ks += 32) {
    const bf16* ap = &Qs[wave * 16 + lr][ks];
    v16bf a = cat8(*(const v8bf*)(ap + 8 * lh), *(const v8bf*)(ap + 16 + 8 * lh));
#pragma unroll
    for (int nt = 0; nt < 4; ++nt) {
      const bf16* bp = &Pt[nt * 16 + lr][ks];
      v16bf bb = cat8(*(const v8bf*)(bp + 8 * lh), *(const v8bf*)(bp + 16 + 8 * lh));
      oacc[nt] = WMMA_BF16(a, bb, oacc[nt]);
    }
  }

#pragma unroll
  for (int nt = 0; nt < 4; ++nt) {
    int dd = nt * 16 + lr;
#pragma unroll
    for (int r = 0; r < 8; ++r) {
      int i = wave * 16 + 8 * lh + r;
      size_t row = (size_t)b * Tt + n * Cc + i;
      attn[row * (size_t)Dd + h * 64 + dd] = (bf16)oacc[nt][r];
    }
  }
}

// ---------------------------------------------------------------------------
// Kernel 5: out = attn @ w_o^T  (M=8192, N=1024, K=1024), fp32 store
// ---------------------------------------------------------------------------
__global__ __launch_bounds__(256)
void out_gemm_kernel(const bf16* __restrict__ A, const float* __restrict__ Wo,
                     float* __restrict__ out) {
  __shared__ bf16 As[128][40];
  __shared__ bf16 Bs[128][40];
  const int n0 = blockIdx.x * 128;
  const int m0 = blockIdx.y * 128;
  const int tid  = threadIdx.x;
  const int lane = tid & 31, wave = tid >> 5;
  const int wm = (wave & 3) * 32;
  const int wn = (wave >> 2) * 64;
  const int lr = lane & 15, lh = lane >> 4;

  v8f acc[2][4] = {};

  for (int kt = 0; kt < 1024; kt += 32) {
    const int ktn = (kt + 32 < 1024) ? (kt + 32) : kt;
    // A tile: pure bf16 copy -> async global->LDS when available
#if ASYNC_OK
#pragma unroll
    for (int c = 0; c < 2; ++c) {
      int q   = tid + 256 * c;
      int row = q >> 2, col = (q & 3) * 8;
      ASYNC_CP16(A + (size_t)(m0 + row) * 1024 + kt + col, &As[row][col]);
    }
#else
#pragma unroll
    for (int c = 0; c < 2; ++c) {
      int q   = tid + 256 * c;
      int row = q >> 2, col = (q & 3) * 8;
      const bf16* ap = A + (size_t)(m0 + row) * 1024 + kt + col;
      *(v8bf*)&As[row][col] = *(const v8bf*)ap;
    }
#endif
#pragma unroll
    for (int c = 0; c < 4; ++c) {               // B: fp32 -> bf16
      int q   = tid + 256 * c;
      int row = q >> 3, col = (q & 7) * 4;
      const float* bp = Wo + (size_t)(n0 + row) * 1024 + kt + col;
      __builtin_prefetch(Wo + (size_t)(n0 + row) * 1024 + ktn + col, 0, 3);
      v4f b4 = *(const v4f*)bp;
#pragma unroll
      for (int j = 0; j < 4; ++j) Bs[row][col + j] = (bf16)b4[j];
    }
#if ASYNC_OK
    wait_async0();
#endif
    __syncthreads();

    v16bf af[2], bfr[4];
#pragma unroll
    for (int mt = 0; mt < 2; ++mt) {
      const bf16* ap = &As[wm + mt * 16 + lr][0];
      af[mt] = cat8(*(const v8bf*)(ap + 8 * lh), *(const v8bf*)(ap + 16 + 8 * lh));
    }
#pragma unroll
    for (int nt = 0; nt < 4; ++nt) {
      const bf16* bp = &Bs[wn + nt * 16 + lr][0];
      bfr[nt] = cat8(*(const v8bf*)(bp + 8 * lh), *(const v8bf*)(bp + 16 + 8 * lh));
    }
#pragma unroll
    for (int mt = 0; mt < 2; ++mt)
#pragma unroll
      for (int nt = 0; nt < 4; ++nt)
        acc[mt][nt] = WMMA_BF16(af[mt], bfr[nt], acc[mt][nt]);
    __syncthreads();
  }

#pragma unroll
  for (int nt = 0; nt < 4; ++nt) {
    int gn = n0 + wn + nt * 16 + lr;
#pragma unroll
    for (int mt = 0; mt < 2; ++mt) {
#pragma unroll
      for (int r = 0; r < 8; ++r) {
        int gm = m0 + wm + mt * 16 + 8 * lh + r;
        out[(size_t)gm * 1024 + gn] = acc[mt][nt][r];
      }
    }
  }
}

// ---------------------------------------------------------------------------
extern "C" void kernel_launch(void* const* d_in, const int* in_sizes, int n_in,
                              void* d_out, int out_size, void* d_ws, size_t ws_size,
                              hipStream_t stream) {
  const float* x     = (const float*)d_in[0];
  const float* w_qkv = (const float*)d_in[1];
  const float* w_o   = (const float*)d_in[2];
  float* out = (float*)d_out;

  char* ws = (char*)d_ws;
  size_t off = 0;
  const size_t QKV_ELEMS = (size_t)Bb * Hh * Tt * DH;        // 8,388,608
  bf16* q_ws = (bf16*)(ws + off); off += QKV_ELEMS * sizeof(bf16);
  bf16* k_ws = (bf16*)(ws + off); off += QKV_ELEMS * sizeof(bf16);
  bf16* v_ws = (bf16*)(ws + off); off += QKV_ELEMS * sizeof(bf16);
  const size_t CKV_ELEMS = (size_t)Bb * Hh * NC * DH * DH;   // 8,388,608
  float* ckv = (float*)(ws + off); off += CKV_ELEMS * sizeof(float);
  float* pre = (float*)(ws + off); off += CKV_ELEMS * sizeof(float);
  bf16* attn = (bf16*)(ws + off); off += (size_t)MROWS * Dd * sizeof(bf16);

  float* fstate = out + (size_t)Bb * Tt * Dd;                // second output

  qkv_gemm_kernel<<<dim3(3072 / 128, MROWS / 128), 256, 0, stream>>>(
      x, w_qkv, q_ws, k_ws, v_ws);
  chunkkv_kernel<<<dim3(NC, Bb * Hh), 128, 0, stream>>>(k_ws, v_ws, ckv);
  prefix_kernel<<<dim3(Bb * Hh), 256, 0, stream>>>(ckv, pre, fstate);
  attn_out_kernel<<<dim3(NC, Bb * Hh), 128, 0, stream>>>(q_ws, k_ws, v_ws, pre, attn);
  out_gemm_kernel<<<dim3(Dd / 128, MROWS / 128), 256, 0, stream>>>(attn, w_o, out);
}
